// Decoder_70145405878759
// MI455X (gfx1250) — matrix operand order repaired and checked by reading, not compiled, not run on hardware
//
#include <hip/hip_runtime.h>
#include <math.h>

// ---------------------------------------------------------------------------
// Types
// ---------------------------------------------------------------------------
typedef _Float16 v8h  __attribute__((ext_vector_type(8)));
typedef _Float16 v16h __attribute__((ext_vector_type(16)));
typedef float    v8f  __attribute__((ext_vector_type(8)));
typedef float    v4f  __attribute__((ext_vector_type(4)));

#define DEV static __device__ __forceinline__

// Problem constants
#define B_  1024
#define S_  100
#define H_  1024
#define E_  1024
#define G_  4096
#define K_  1024
#define KC_ 32          // K_/32 wmma k-steps

DEV float sigf(float x) { return 1.0f / (1.0f + expf(-x)); }

// Load one WMMA operand fragment (16 halfs/lane) from fragment-major storage:
// layout [tile][kc][piece(2)][lane(32)][8 halfs]; piece0 at +0, piece1 at +256h.
DEV v16h ldfrag(const _Float16* p) {
  v8h lo = *(const v8h*)p;
  v8h hi = *(const v8h*)(p + 256);
  return __builtin_shufflevector(lo, hi, 0,1,2,3,4,5,6,7,8,9,10,11,12,13,14,15);
}

DEV v8f wmma16(v16h a, v16h b, v8f c) {
  return __builtin_amdgcn_wmma_f32_16x16x32_f16(
      /*neg_a=*/false, a, /*neg_b=*/false, b,
      /*c_mod=*/(short)0, c, /*reuse_a=*/false, /*reuse_b=*/false);
}

// A-matrix fragment run index for row r, k-run j0 (j0 % 8 == 0).
// A 16x32 f16 layout: lanes<16: K 0..7 (piece0), 16..23 (piece1); lanes>=16: 8..15, 24..31.
DEV size_t permA_run(int r, int j0) {
  size_t t     = (size_t)(r >> 4);
  int    kc    = j0 >> 5;
  int    piece = (j0 >> 4) & 1;
  int    l     = (r & 15) + (((j0 >> 3) & 1) << 4);
  return (((t * KC_ + kc) * 2 + piece) << 5) + l;   // *8 halfs = element offset
}

// ---------------------------------------------------------------------------
// Prep: weights f32 -> f16 in B-matrix fragment order
// B 32x16 f16 layout: lanes<16 hold K 0..7,16..23 (col = lane), lanes>=16 hold K 8..15,24..31.
// ---------------------------------------------------------------------------
struct Ptr10 { const float* p[10]; };
struct Ptr4  { const float* p[4];  };

__global__ __launch_bounds__(256) void k_prep_w(Ptr10 src, _Float16* dst) {
  int gid = blockIdx.x * 256 + threadIdx.x;          // one 8-half run
  int m   = gid >> 19;                               // matrix id (runs/matrix = 2^19)
  int rid = gid & ((1 << 19) - 1);
  int l     = rid & 31;
  int piece = (rid >> 5) & 1;
  int kc    = (rid >> 6) & 31;
  int t     = rid >> 11;                             // 0..255
  int n = t * 16 + (l & 15);
  int k = kc * 32 + (piece << 4) + ((l >> 4) << 3);
  const float* s = src.p[m] + (size_t)n * K_ + k;
  v8h v;
#pragma unroll
  for (int e = 0; e < 8; ++e) v[e] = (_Float16)s[e];
  *(v8h*)(dst + (size_t)m * ((size_t)G_ * K_) + (size_t)rid * 8) = v;
}

// Prep: 4 activation matrices (1024x1024) f32 -> f16 in A-matrix fragment order
__global__ __launch_bounds__(256) void k_prep_a4(Ptr4 src, _Float16* dst) {
  int gid = blockIdx.x * 256 + threadIdx.x;          // runs/matrix = 2^17
  int m   = gid >> 17;
  int rid = gid & ((1 << 17) - 1);
  int l     = rid & 31;
  int piece = (rid >> 5) & 1;
  int kc    = (rid >> 6) & 31;
  int t     = rid >> 11;                             // 0..63
  int r = t * 16 + (l & 15);
  int k = kc * 32 + (piece << 4) + ((l >> 4) << 3);
  const float* s = src.p[m] + (size_t)r * K_ + k;
  v8h v;
#pragma unroll
  for (int e = 0; e < 8; ++e) v[e] = (_Float16)s[e];
  *(v8h*)(dst + (size_t)m * ((size_t)B_ * K_) + (size_t)rid * 8) = v;
}

// Embedding gather -> A-fragment f16
__global__ __launch_bounds__(256) void k_gather(const int* __restrict__ prev_y,
                                                const float* __restrict__ emb,
                                                _Float16* __restrict__ dst) {
  int rid = blockIdx.x * 256 + threadIdx.x;          // 2^17 runs
  int l     = rid & 31;
  int piece = (rid >> 5) & 1;
  int kc    = (rid >> 6) & 31;
  int t     = rid >> 11;
  int b = t * 16 + (l & 15);
  int k = kc * 32 + (piece << 4) + ((l >> 4) << 3);
  int row = prev_y[b];
  const float* s = emb + (size_t)row * E_ + k;
  v8h v;
#pragma unroll
  for (int e = 0; e < 8; ++e) v[e] = (_Float16)s[e];
  *(v8h*)(dst + (size_t)rid * 8) = v;
}

// ---------------------------------------------------------------------------
// Fused multi-source GEMM: C(1024x4096,f32) = sum_p A_p(1024xK) @ W_p(4096xK)^T
// Block 256 thr = 8 waves; block tile 64x128; wave tile 32x32 (2x2 wmma tiles).
// ---------------------------------------------------------------------------
struct GemmArgs { const _Float16* a[4]; const _Float16* w[4]; int np; };

__global__ __launch_bounds__(256) void k_gemm(GemmArgs g, float* __restrict__ C) {
  const int lane = threadIdx.x & 31;
  const int wave = threadIdx.x >> 5;
  const int wm = wave & 1;        // 2 waves along M
  const int wn = wave >> 1;       // 4 waves along N
  const int tm0 = blockIdx.y * 4 + wm * 2;   // 16-row tile index (0..63)
  const int tn0 = blockIdx.x * 8 + wn * 2;   // 16-col tile index (0..255)

  v8f c00 = {}, c01 = {}, c10 = {}, c11 = {};

  for (int p = 0; p < g.np; ++p) {
    const _Float16* a0p = g.a[p] + ((size_t)tm0 * KC_) * 512 + (size_t)lane * 8;
    const _Float16* a1p = a0p + (size_t)KC_ * 512;
    const _Float16* b0p = g.w[p] + ((size_t)tn0 * KC_) * 512 + (size_t)lane * 8;
    const _Float16* b1p = b0p + (size_t)KC_ * 512;
#pragma unroll 2
    for (int kc = 0; kc < KC_; ++kc) {
      v16h a0 = ldfrag(a0p);
      v16h a1 = ldfrag(a1p);
      v16h b0 = ldfrag(b0p);
      v16h b1 = ldfrag(b1p);
      a0p += 512; a1p += 512; b0p += 512; b1p += 512;
      c00 = wmma16(a0, b0, c00);
      c01 = wmma16(a0, b1, c01);
      c10 = wmma16(a1, b0, c10);
      c11 = wmma16(a1, b1, c11);
    }
  }

  // C/D layout: VGPR r: lanes0-15 -> M=r, lanes16-31 -> M=8+r; N = lane&15
  const int col0 = tn0 * 16 + (lane & 15);
  const int row0 = tm0 * 16 + ((lane >> 4) << 3);
#pragma unroll
  for (int r = 0; r < 8; ++r) {
    C[(size_t)(row0 + r)      * G_ + col0]      = c00[r];
    C[(size_t)(row0 + r)      * G_ + col0 + 16] = c01[r];
    C[(size_t)(row0 + 16 + r) * G_ + col0]      = c10[r];
    C[(size_t)(row0 + 16 + r) * G_ + col0 + 16] = c11[r];
  }
}

// ---------------------------------------------------------------------------
// LSTM elementwise: gates(B,4G) + bih + bhh, c_in -> h_out (plain f32)
// ---------------------------------------------------------------------------
__global__ __launch_bounds__(256) void k_lstm(const float* __restrict__ gates,
                                              const float* __restrict__ bih,
                                              const float* __restrict__ bhh,
                                              const float* __restrict__ c_in,
                                              float* __restrict__ h_out) {
  int idx = blockIdx.x * 256 + threadIdx.x;   // B_*H_
  int b = idx >> 10, j = idx & (H_ - 1);
  const float* gr = gates + (size_t)b * G_;
  float gi = gr[j]        + bih[j]        + bhh[j];
  float gf = gr[j + 1024] + bih[j + 1024] + bhh[j + 1024];
  float gg = gr[j + 2048] + bih[j + 2048] + bhh[j + 2048];
  float go = gr[j + 3072] + bih[j + 3072] + bhh[j + 3072];
  float c  = c_in[(size_t)b * H_ + j];
  float c2 = sigf(gf) * c + sigf(gi) * tanhf(gg);
  h_out[(size_t)b * H_ + j] = sigf(go) * tanhf(c2);
}

// Final gate: gates + b3, c3 -> hx (d_out, plain f32)
__global__ __launch_bounds__(256) void k_final(const float* __restrict__ gates,
                                               const float* __restrict__ b3,
                                               const float* __restrict__ c3,
                                               float* __restrict__ out) {
  int idx = blockIdx.x * 256 + threadIdx.x;
  int b = idx >> 10, j = idx & (H_ - 1);
  const float* gr = gates + (size_t)b * G_;
  float gi = gr[j]        + b3[j];
  float gf = gr[j + 1024] + b3[j + 1024];
  float gg = gr[j + 2048] + b3[j + 2048];
  float go = gr[j + 3072] + b3[j + 3072];
  float c  = c3[(size_t)b * H_ + j];
  float cx = sigf(gf) * c + sigf(gi) * tanhf(gg);
  out[(size_t)b * H_ + j] = sigf(go) * tanhf(cx);
}

// ---------------------------------------------------------------------------
// LayerNorm per row -> A-fragment f16.  128 threads/row, 8 elems/thread.
// ---------------------------------------------------------------------------
__global__ __launch_bounds__(128) void k_ln(const float* __restrict__ x,
                                            const float* __restrict__ gam,
                                            const float* __restrict__ bet,
                                            _Float16* __restrict__ dstp) {
  __shared__ float red[8];
  int b = blockIdx.x, tid = threadIdx.x;
  int lane = tid & 31, wave = tid >> 5;
  int j0 = tid * 8;
  const float* xr = x + (size_t)b * H_;
  float v[8]; float s = 0.f;
#pragma unroll
  for (int e = 0; e < 8; ++e) { v[e] = xr[j0 + e]; s += v[e]; }
#pragma unroll
  for (int off = 16; off; off >>= 1) s += __shfl_xor(s, off, 32);
  if (lane == 0) red[wave] = s;
  __syncthreads();
  float mean = (red[0] + red[1] + red[2] + red[3]) * (1.0f / H_);
  float s2 = 0.f;
#pragma unroll
  for (int e = 0; e < 8; ++e) { float d = v[e] - mean; s2 += d * d; }
#pragma unroll
  for (int off = 16; off; off >>= 1) s2 += __shfl_xor(s2, off, 32);
  if (lane == 0) red[4 + wave] = s2;
  __syncthreads();
  float var  = (red[4] + red[5] + red[6] + red[7]) * (1.0f / H_);
  float rstd = rsqrtf(var + 1e-5f);
  v8h o;
#pragma unroll
  for (int e = 0; e < 8; ++e)
    o[e] = (_Float16)(gam[j0 + e] * (v[e] - mean) * rstd + bet[j0 + e]);
  *(v8h*)(dstp + permA_run(b, j0) * 8) = o;
}

// ---------------------------------------------------------------------------
// Single-pass attention (online softmax, flash style). One block per batch row.
// 8 waves; wave w streams s = w, w+8, ...  ctx is read from HBM exactly once.
// Per lane: owns h in {lane*4 + 128*i + q : i<8, q<4} (32 f32 accum + row regs).
// Merge across waves via tiny LDS reduction + ds_add_f32 atomics, then write
// the normalized result straight into the WMMA A-fragment layout (f16).
// ---------------------------------------------------------------------------
__global__ __launch_bounds__(256) void k_attn_flash(const float* __restrict__ ctx,
                                                    const float* __restrict__ h0,
                                                    const unsigned char* __restrict__ mask,
                                                    _Float16* __restrict__ attnp) {
  __shared__ float h0l[H_];
  __shared__ float accs[H_];
  __shared__ float red_m[8];
  __shared__ float red_z[8];
  int b = blockIdx.x, tid = threadIdx.x, lane = tid & 31, wave = tid >> 5;

  for (int j = tid; j < H_; j += 256) {
    h0l[j]  = h0[(size_t)b * H_ + j];
    accs[j] = 0.f;
  }
  __syncthreads();

  v4f hv[8];
#pragma unroll
  for (int i = 0; i < 8; ++i) hv[i] = *(const v4f*)(h0l + lane * 4 + 128 * i);

  v4f acc[8];
#pragma unroll
  for (int i = 0; i < 8; ++i) acc[i] = (v4f){0.f, 0.f, 0.f, 0.f};
  float m = -3.0e38f, z = 0.f;

  for (int s = wave; s < S_; s += 8) {
    const float* cr = ctx + ((size_t)b * S_ + s) * H_;
    v4f v[8];
    float d = 0.f;
#pragma unroll
    for (int i = 0; i < 8; ++i) {
      v[i] = *(const v4f*)(cr + lane * 4 + 128 * i);
      d += v[i][0] * hv[i][0] + v[i][1] * hv[i][1] +
           v[i][2] * hv[i][2] + v[i][3] * hv[i][3];
    }
#pragma unroll
    for (int off = 16; off; off >>= 1) d += __shfl_xor(d, off, 32);
    float score = mask[(size_t)b * S_ + s] ? -1.0e9f : d;
    float m2 = fmaxf(m, score);
    float sc = expf(m - m2);        // underflows to 0 on first iteration
    float p  = expf(score - m2);
    z = z * sc + p;
#pragma unroll
    for (int i = 0; i < 8; ++i) {
#pragma unroll
      for (int q = 0; q < 4; ++q) acc[i][q] = acc[i][q] * sc + p * v[i][q];
    }
    m = m2;
  }

  if (lane == 0) { red_m[wave] = m; red_z[wave] = z; }
  __syncthreads();

  float M = red_m[0];
#pragma unroll
  for (int w = 1; w < 8; ++w) M = fmaxf(M, red_m[w]);
  float zt = 0.f;
#pragma unroll
  for (int w = 0; w < 8; ++w) zt += red_z[w] * expf(red_m[w] - M);
  float f = expf(m - M);            // per-wave rescale factor

#pragma unroll
  for (int i = 0; i < 8; ++i) {
    int h = lane * 4 + 128 * i;
#pragma unroll
    for (int q = 0; q < 4; ++q) atomicAdd(&accs[h + q], acc[i][q] * f);
  }
  __syncthreads();

  if (tid < 128) {
    int j0 = tid * 8;
    float inv = 1.0f / zt;
    v8h o;
#pragma unroll
    for (int e = 0; e < 8; ++e) o[e] = (_Float16)(accs[j0 + e] * inv);
    *(v8h*)(attnp + permA_run(b, j0) * 8) = o;
  }
}

// ---------------------------------------------------------------------------
// Host launch
// ---------------------------------------------------------------------------
extern "C" void kernel_launch(void* const* d_in, const int* in_sizes, int n_in,
                              void* d_out, int out_size, void* d_ws, size_t ws_size,
                              hipStream_t stream) {
  (void)in_sizes; (void)n_in; (void)out_size; (void)ws_size;
  const int*           prev_y = (const int*)d_in[0];
  const unsigned char* mask   = (const unsigned char*)d_in[1];
  const float*         ctx    = (const float*)d_in[2];
  const float*         emb    = (const float*)d_in[3];
  const float* Wih[3] = {(const float*)d_in[4],  (const float*)d_in[10], (const float*)d_in[16]};
  const float* Whh[3] = {(const float*)d_in[5],  (const float*)d_in[11], (const float*)d_in[17]};
  const float* bih[3] = {(const float*)d_in[6],  (const float*)d_in[12], (const float*)d_in[18]};
  const float* bhh[3] = {(const float*)d_in[7],  (const float*)d_in[13], (const float*)d_in[19]};
  const float* gam[3] = {(const float*)d_in[8],  (const float*)d_in[14], (const float*)d_in[20]};
  const float* bet[3] = {(const float*)d_in[9],  (const float*)d_in[15], (const float*)d_in[21]};
  const float* h_in[4] = {(const float*)d_in[22], (const float*)d_in[24],
                          (const float*)d_in[26], (const float*)d_in[28]};
  const float* c_in[4] = {(const float*)d_in[23], (const float*)d_in[25],
                          (const float*)d_in[27], (const float*)d_in[29]};
  const float* Wx3 = (const float*)d_in[30];
  const float* Wh3 = (const float*)d_in[31];
  const float* Wc3 = (const float*)d_in[32];
  const float* Ws3 = (const float*)d_in[33];
  const float* b3  = (const float*)d_in[34];

  // Workspace carve-up
  char* ws = (char*)d_ws;
  const size_t WSLOT = (size_t)G_ * K_;          // halfs per weight slot
  const size_t ASLOT = (size_t)B_ * K_;          // halfs per activation slot
  _Float16* WP    = (_Float16*)ws;                          ws += 10 * WSLOT * 2;
  _Float16* XP    = (_Float16*)ws;                          ws += ASLOT * 2;
  _Float16* HP    = (_Float16*)ws;                          ws += 4 * ASLOT * 2;  // h0..h3
  _Float16* LNP0  = (_Float16*)ws;                          ws += ASLOT * 2;
  _Float16* LNP1  = (_Float16*)ws;                          ws += ASLOT * 2;
  _Float16* LNP2  = (_Float16*)ws;                          ws += ASLOT * 2;
  _Float16* ATTNP = (_Float16*)ws;                          ws += ASLOT * 2;
  float*    GATES = (float*)ws;                             ws += (size_t)B_ * G_ * 4;
  float*    H0P   = (float*)ws;                             ws += (size_t)B_ * H_ * 4; // layer0 h (plain)
  float*    HT    = (float*)ws;                             ws += (size_t)B_ * H_ * 4; // layer1/2 h (plain)

  // ---- prep: weights + h-states + embedding gather (fragment layouts)
  Ptr10 pw = {{Wih[0], Whh[0], Wih[1], Whh[1], Wih[2], Whh[2], Wx3, Wh3, Wc3, Ws3}};
  k_prep_w<<<20480, 256, 0, stream>>>(pw, WP);
  Ptr4 ph = {{h_in[0], h_in[1], h_in[2], h_in[3]}};
  k_prep_a4<<<2048, 256, 0, stream>>>(ph, HP);
  k_gather<<<512, 256, 0, stream>>>(prev_y, emb, XP);

  const dim3 gGrid(32, 16);
  _Float16* WihP[3] = {WP + 0 * WSLOT, WP + 2 * WSLOT, WP + 4 * WSLOT};
  _Float16* WhhP[3] = {WP + 1 * WSLOT, WP + 3 * WSLOT, WP + 5 * WSLOT};

  // ---- layer 0
  {
    GemmArgs g = {{XP, HP + 0 * ASLOT, nullptr, nullptr},
                  {WihP[0], WhhP[0], nullptr, nullptr}, 2};
    k_gemm<<<gGrid, 256, 0, stream>>>(g, GATES);
  }
  k_lstm<<<4096, 256, 0, stream>>>(GATES, bih[0], bhh[0], c_in[0], H0P);
  k_ln<<<1024, 128, 0, stream>>>(H0P, gam[0], bet[0], LNP0);

  // ---- attention (depends only on layer-0 h; single pass over ctx)
  k_attn_flash<<<1024, 256, 0, stream>>>(ctx, H0P, mask, ATTNP);

  // ---- layer 1
  {
    GemmArgs g = {{LNP0, HP + 1 * ASLOT, nullptr, nullptr},
                  {WihP[1], WhhP[1], nullptr, nullptr}, 2};
    k_gemm<<<gGrid, 256, 0, stream>>>(g, GATES);
  }
  k_lstm<<<4096, 256, 0, stream>>>(GATES, bih[1], bhh[1], c_in[1], HT);
  k_ln<<<1024, 128, 0, stream>>>(HT, gam[1], bet[1], LNP1);

  // ---- layer 2
  {
    GemmArgs g = {{LNP1, HP + 2 * ASLOT, nullptr, nullptr},
                  {WihP[2], WhhP[2], nullptr, nullptr}, 2};
    k_gemm<<<gGrid, 256, 0, stream>>>(g, GATES);
  }
  k_lstm<<<4096, 256, 0, stream>>>(GATES, bih[2], bhh[2], c_in[2], HT);
  k_ln<<<1024, 128, 0, stream>>>(HT, gam[2], bet[2], LNP2);

  // ---- final 4-way GEMM gate
  {
    GemmArgs g = {{XP, HP + 3 * ASLOT, ATTNP, LNP2},
                  {WP + 6 * WSLOT, WP + 7 * WSLOT, WP + 8 * WSLOT, WP + 9 * WSLOT}, 4};
    k_gemm<<<gGrid, 256, 0, stream>>>(g, GATES);
  }
  k_final<<<4096, 256, 0, stream>>>(GATES, b3, c_in[3], (float*)d_out);
}
